// PointNetEstimationV2_28166395527718
// MI455X (gfx1250) — matrix operand order repaired and checked by reading, not compiled, not run on hardware
//
#include <hip/hip_runtime.h>
#include <hip/hip_bf16.h>

// ---------------------------------------------------------------------------
// PointNet box-estimation for MI455X (gfx1250, wave32, WMMA).
// All dense MLP/FC math runs through v_wmma_f32_16x16x32_f16 (f16 in, f32 acc).
// Each wave owns one 16-row M tile and NT 16-col N tiles: the A fragment is
// loaded once per K-step and reused across NT WMMA ops (cuts X traffic NT x).
// FPS / ball-query use LDS staging; BN batch statistics via in-GEMM atomics.
// ---------------------------------------------------------------------------

typedef __attribute__((ext_vector_type(16))) _Float16 v16h;
typedef __attribute__((ext_vector_type(8)))  _Float16 v8h;
typedef __attribute__((ext_vector_type(8)))  float    v8f;

// ---------------------------------------------------------------------------
// Farthest point sampling: one block per batch, points cached in LDS.
// Layout-generic via strides (stage1: pts is (B,3,N); stage2: (B,N,3)).
// ---------------------------------------------------------------------------
__global__ void fps_kernel(const float* __restrict__ xyz, int N, int npoint,
                           int strideC, int strideP, int batchStride,
                           int* __restrict__ out) {
    __shared__ float sx[1024], sy[1024], sz[1024];
    __shared__ float rv[256];
    __shared__ int   ri[256];
    const int b = blockIdx.x, tid = threadIdx.x, T = blockDim.x;
    const float* base = xyz + (size_t)b * batchStride;
    for (int p = tid; p < N; p += T) {
        sx[p] = base[0 * strideC + p * strideP];
        sy[p] = base[1 * strideC + p * strideP];
        sz[p] = base[2 * strideC + p * strideP];
    }
    __syncthreads();
    float dist[4];
    for (int j = 0; j < 4; ++j) dist[j] = 1e10f;
    int far = 0;
    for (int it = 0; it < npoint; ++it) {
        if (tid == 0) out[b * npoint + it] = far;
        const float cx = sx[far], cy = sy[far], cz = sz[far];
        float best = -1.0f; int bi = 0;
        for (int j = 0; j < 4; ++j) {
            int p = tid + j * T;
            if (p < N) {
                float dx = sx[p] - cx, dy = sy[p] - cy, dz = sz[p] - cz;
                float d = dx * dx + dy * dy + dz * dz;
                if (d < dist[j]) dist[j] = d;
                if (dist[j] > best) { best = dist[j]; bi = p; }
            }
        }
        rv[tid] = best; ri[tid] = bi;
        __syncthreads();
        for (int s = T >> 1; s > 0; s >>= 1) {
            if (tid < s) {
                float ov = rv[tid + s]; int oi = ri[tid + s];
                if (ov > rv[tid] || (ov == rv[tid] && oi < ri[tid])) {
                    rv[tid] = ov; ri[tid] = oi;
                }
            }
            __syncthreads();
        }
        far = ri[0];
        __syncthreads();
    }
}

// new_xyz[(b,s,c)] = xyz[b, fidx[b,s]] (strided source)
__global__ void gather_centers(const float* __restrict__ xyz,
                               const int* __restrict__ fidx,
                               float* __restrict__ newxyz, int npoint,
                               int strideC, int strideP, int batchStride, int total) {
    int i = blockIdx.x * blockDim.x + threadIdx.x;
    if (i >= total) return;
    int c = i % 3; int bs = i / 3;
    int s = bs % npoint; int b = bs / npoint;
    int p = fidx[b * npoint + s];
    newxyz[i] = xyz[(size_t)b * batchStride + c * strideC + p * strideP];
}

// Ball query: first (up to) 32 in-radius indices in ascending order, padded
// with the first hit (matches reference top_k-of-negated-indices trick).
__global__ void ball_query_kernel(const float* __restrict__ xyz,
                                  const float* __restrict__ newxyz,
                                  int N, int S, float r2,
                                  int strideC, int strideP, int batchStride,
                                  int* __restrict__ idx) {
    __shared__ float sx[1024], sy[1024], sz[1024];
    const int b = blockIdx.x, tid = threadIdx.x, T = blockDim.x;
    const float* base = xyz + (size_t)b * batchStride;
    for (int p = tid; p < N; p += T) {
        sx[p] = base[0 * strideC + p * strideP];
        sy[p] = base[1 * strideC + p * strideP];
        sz[p] = base[2 * strideC + p * strideP];
    }
    __syncthreads();
    if (tid >= S) return;
    const float* c = newxyz + ((size_t)b * S + tid) * 3;
    const float cx = c[0], cy = c[1], cz = c[2];
    int* o = idx + ((size_t)b * S + tid) * 32;
    int cnt = 0, first = 0;
    for (int p = 0; p < N && cnt < 32; ++p) {
        float dx = sx[p] - cx, dy = sy[p] - cy, dz = sz[p] - cz;
        if (dx * dx + dy * dy + dz * dz <= r2) {
            if (cnt == 0) first = p;
            o[cnt++] = p;
        }
    }
    for (; cnt < 32; ++cnt) o[cnt] = first;
}

// ---------------------------------------------------------------------------
// Gather grouped features into padded f16 matrices (zero-padded K columns).
// ---------------------------------------------------------------------------
__global__ void gather_sa1(const float* __restrict__ pts,
                           const float* __restrict__ nx1,
                           const int* __restrict__ idx,
                           _Float16* __restrict__ X, int total) {
    int i = blockIdx.x * blockDim.x + threadIdx.x;
    if (i >= total) return;
    int c = i & 31; int row = i >> 5;           // Cpad = 32
    int k = row & 31; int bs = row >> 5;        // bs = b*128 + s
    float v = 0.0f;
    if (c < 3) {
        int p = idx[(size_t)bs * 32 + k];
        int b = bs >> 7;
        v = pts[(size_t)b * 3072 + c * 1024 + p] - nx1[(size_t)bs * 3 + c];
    }
    X[i] = (_Float16)v;
}

__global__ void gather_sa2(const float* __restrict__ nx1,
                           const float* __restrict__ nx2,
                           const float* __restrict__ pooled1,
                           const int* __restrict__ idx,
                           _Float16* __restrict__ X, int total) {
    int i = blockIdx.x * blockDim.x + threadIdx.x;
    if (i >= total) return;
    int c = i % 96; int row = i / 96;           // Cpad = 96
    int k = row & 31; int bs = row >> 5;        // bs = b*32 + s
    int b = bs >> 5;
    float v = 0.0f;
    int p = idx[(size_t)bs * 32 + k];
    if (c < 3)       v = nx1[((size_t)b * 128 + p) * 3 + c] - nx2[(size_t)bs * 3 + c];
    else if (c < 67) v = pooled1[((size_t)b * 128 + p) * 64 + (c - 3)];
    X[i] = (_Float16)v;
}

__global__ void gather_sa3(const float* __restrict__ nx2,
                           const float* __restrict__ pooled2,
                           _Float16* __restrict__ X, int total) {
    int i = blockIdx.x * blockDim.x + threadIdx.x;
    if (i >= total) return;
    int c = i % 160; int row = i / 160;         // row = b*32 + k, Cpad = 160
    float v = 0.0f;
    if (c < 3)        v = nx2[(size_t)row * 3 + c];
    else if (c < 131) v = pooled2[(size_t)row * 128 + (c - 3)];
    X[i] = (_Float16)v;
}

// ---------------------------------------------------------------------------
// WMMA GEMM: Y[M][CoutPad] = X[M][Kc] @ W^T + bias
// One wave per 16-row M tile; NT consecutive 16-col N tiles per wave so the A
// fragment is loaded once and fed to NT v_wmma ops per K-step.
// Fragment loads match the CDNA5 16-bit A/B VGPR layouts directly from global.
// Optionally accumulates per-channel sum / sum-of-squares for batchnorm.
// ---------------------------------------------------------------------------
template <int NT>
__global__ void __launch_bounds__(32)
gemm_wmma_nt(const _Float16* __restrict__ X, const _Float16* __restrict__ W,
             const float* __restrict__ bias, _Float16* __restrict__ Yh,
             float* __restrict__ Yf, int Kc, int CoutPad,
             float* __restrict__ ssum, float* __restrict__ ssq) {
    const int lane  = threadIdx.x;
    const int mlane = lane & 15;
    const int hi    = lane >> 4;                 // lane group 0/1
    const size_t m0 = (size_t)blockIdx.x * 16;
    const int    n0 = blockIdx.y * (NT * 16);

    v8f acc[NT];
    v8f zero = {};
    for (int t = 0; t < NT; ++t) acc[t] = zero;

    // A: 16x32 f16. lanes 0-15 hold K 0-7 / 16-23, lanes 16-31 K 8-15 / 24-31.
    const _Float16* xrow  = X + (m0 + mlane) * (size_t)Kc + hi * 8;
    // B: 32x16 f16. lane n%16 = column, K 0-7 / 16-23 (lanes 0-15), 8-15 / 24-31 (lanes 16-31).
    const _Float16* wbase = W + (size_t)(n0 + mlane) * Kc + hi * 8;

    for (int k0 = 0; k0 < Kc; k0 += 32) {
        if (k0 + 32 < Kc)
            __builtin_prefetch(xrow + k0 + 32, 0, 1);   // global_prefetch_b8
        v8h alo = *(const v8h*)(xrow + k0);
        v8h ahi = *(const v8h*)(xrow + k0 + 16);
        v16h a = __builtin_shufflevector(alo, ahi,
                 0, 1, 2, 3, 4, 5, 6, 7, 8, 9, 10, 11, 12, 13, 14, 15);
        for (int t = 0; t < NT; ++t) {
            v8h blo = *(const v8h*)(wbase + (size_t)t * 16 * Kc + k0);
            v8h bhi = *(const v8h*)(wbase + (size_t)t * 16 * Kc + k0 + 16);
            v16h b = __builtin_shufflevector(blo, bhi,
                     0, 1, 2, 3, 4, 5, 6, 7, 8, 9, 10, 11, 12, 13, 14, 15);
            acc[t] = __builtin_amdgcn_wmma_f32_16x16x32_f16(
                        false, a, false, b, (short)0, acc[t], false, false);
        }
    }

    for (int t = 0; t < NT; ++t) {
        const int n = n0 + t * 16 + mlane;
        const float bv = bias[n];
        float s = 0.0f, s2 = 0.0f;
        for (int r = 0; r < 8; ++r) {
            float v = acc[t][r] + bv;
            size_t row = m0 + (size_t)(hi * 8 + r);
            if (Yh) Yh[row * CoutPad + n] = (_Float16)v;
            if (Yf) Yf[row * CoutPad + n] = v;
            s += v; s2 += v * v;
        }
        if (ssum) {
            s  += __shfl_xor(s, 16, 32);
            s2 += __shfl_xor(s2, 16, 32);
            if (hi == 0) {
                atomicAdd(&ssum[n], s);
                atomicAdd(&ssq[n], s2);
            }
        }
    }
}

__global__ void zero_f32(float* __restrict__ p, int n) {
    int i = blockIdx.x * blockDim.x + threadIdx.x;
    if (i < n) p[i] = 0.0f;
}

// scale = gamma * rsqrt(var + eps); shift = beta - mean*scale (padded -> 0)
__global__ void bn_finalize(const float* __restrict__ ssum, const float* __restrict__ ssq,
                            const float* __restrict__ gamma, const float* __restrict__ beta,
                            float* __restrict__ scale, float* __restrict__ shift,
                            int C, float inv_cnt) {
    int i = threadIdx.x;
    float sc = 0.0f, sh = 0.0f;
    if (i < C) {
        float mean = ssum[i] * inv_cnt;
        float var  = ssq[i] * inv_cnt - mean * mean;
        sc = gamma[i] * rsqrtf(var + 1e-5f);
        sh = beta[i] - mean * sc;
    }
    scale[i] = sc; shift[i] = sh;
}

// In-place BN + ReLU on f16 activations (padded channels stay zero).
__global__ void bnrelu_kernel(_Float16* __restrict__ Y,
                              const float* __restrict__ scale,
                              const float* __restrict__ shift,
                              int Cpad, size_t total) {
    size_t i = (size_t)blockIdx.x * blockDim.x + threadIdx.x;
    if (i >= total) return;
    int ch = (int)(i % Cpad);
    float v = (float)Y[i] * scale[ch] + shift[ch];
    Y[i] = (_Float16)fmaxf(v, 0.0f);
}

// BN + ReLU + max over the K grouped points: max_k relu(v_k) == max(0, max_k v_k)
__global__ void bnrelu_maxk(const _Float16* __restrict__ Y,
                            const float* __restrict__ scale,
                            const float* __restrict__ shift,
                            float* __restrict__ pooled,
                            int K, int Cpad, int C, int total) {
    int i = blockIdx.x * blockDim.x + threadIdx.x;
    if (i >= total) return;
    int ch = i % C; int bs = i / C;
    const float sc = scale[ch], sh = shift[ch];
    float m = 0.0f;
    const _Float16* p = Y + (size_t)bs * K * Cpad + ch;
    for (int k = 0; k < K; ++k) {
        float v = (float)p[(size_t)k * Cpad] * sc + sh;
        m = fmaxf(m, v);
    }
    pooled[i] = m;
}

// x = concat(global_feat (B,256), one_hot (B,2)) -> f16 padded to 288
__global__ void build_fc_input(const float* __restrict__ g,
                               const float* __restrict__ onehot,
                               _Float16* __restrict__ X, int total) {
    int i = blockIdx.x * blockDim.x + threadIdx.x;
    if (i >= total) return;
    int c = i % 288; int b = i / 288;
    float v = 0.0f;
    if (c < 256)      v = g[b * 256 + c];
    else if (c < 258) v = onehot[b * 2 + (c - 256)];
    X[i] = (_Float16)v;
}

__global__ void final_copy(const float* __restrict__ Yf, float* __restrict__ out, int total) {
    int i = blockIdx.x * blockDim.x + threadIdx.x;
    if (i >= total) return;
    int j = i % 39; int b = i / 39;
    out[i] = Yf[b * 48 + j];
}

// f32 weights (Cout,Cin) -> padded f16 (CoutPad,Kc); bias -> padded f32.
__global__ void prep_weight(const float* __restrict__ W, const float* __restrict__ bias,
                            _Float16* __restrict__ Wh, float* __restrict__ bp,
                            int Cout, int Cin, int Kc, int total) {
    int i = blockIdx.x * blockDim.x + threadIdx.x;
    if (i >= total) return;
    int k = i % Kc; int o = i / Kc;
    float v = (o < Cout && k < Cin) ? W[o * Cin + k] : 0.0f;
    Wh[i] = (_Float16)v;
    if (k == 0) bp[o] = (o < Cout) ? bias[o] : 0.0f;
}

// ---------------------------------------------------------------------------
static inline unsigned gb(size_t n, unsigned t) { return (unsigned)((n + t - 1) / t); }

extern "C" void kernel_launch(void* const* d_in, const int* in_sizes, int n_in,
                              void* d_out, int out_size, void* d_ws, size_t ws_size,
                              hipStream_t stream) {
    (void)in_sizes; (void)n_in; (void)out_size; (void)ws_size;

    const float* pts    = (const float*)d_in[0];   // (256, 3, 1024)
    const float* onehot = (const float*)d_in[1];   // (256, 2)

    char* ws = (char*)d_ws;
    size_t off = 0;
    auto alloc = [&](size_t bytes) -> void* {
        void* p = ws + off;
        off += (bytes + 255) & ~(size_t)255;
        return p;
    };

    const int M1 = 256 * 128 * 32;   // 1,048,576
    const int M2 = 256 * 32 * 32;    //   262,144
    const int M3 = 256 * 32;         //     8,192

    int*   fidx1   = (int*)  alloc((size_t)256 * 128 * 4);
    int*   fidx2   = (int*)  alloc((size_t)256 * 32 * 4);
    float* nx1     = (float*)alloc((size_t)256 * 128 * 3 * 4);
    float* nx2     = (float*)alloc((size_t)256 * 32 * 3 * 4);
    int*   idx1    = (int*)  alloc((size_t)256 * 128 * 32 * 4);
    int*   idx2    = (int*)  alloc((size_t)256 * 32 * 32 * 4);
    float* pooled1 = (float*)alloc((size_t)256 * 128 * 64 * 4);
    float* pooled2 = (float*)alloc((size_t)256 * 32 * 128 * 4);
    float* pooled3 = (float*)alloc((size_t)256 * 256 * 4);
    float* ssum    = (float*)alloc(256 * 4);
    float* ssq     = (float*)alloc(256 * 4);
    float* scaleb  = (float*)alloc(256 * 4);
    float* shiftb  = (float*)alloc(256 * 4);
    float* fcOut   = (float*)alloc((size_t)256 * 48 * 4);
    _Float16* fcX  = (_Float16*)alloc((size_t)256 * 288 * 2);

    // layer table: {weight d_in index, Cout, Cin, CoutPad, Kc}
    struct LD { int widx, Cout, Cin, CoutPad, Kc; };
    const LD L[12] = {
        { 2,  32,   3,  32,  32}, { 6,  32,  32,  32,  32}, {10,  64,  32,  64,  32},
        {14,  64,  67,  64,  96}, {18,  64,  64,  64,  64}, {22, 128,  64, 128,  64},
        {26, 128, 131, 128, 160}, {30, 128, 128, 128, 128}, {34, 256, 128, 256, 128},
        {38, 256, 258, 256, 288}, {42, 128, 256, 128, 256}, {46,  39, 128,  48, 128}
    };
    _Float16* Wh[12]; float* Bp[12];
    for (int l = 0; l < 12; ++l) {
        Wh[l] = (_Float16*)alloc((size_t)L[l].CoutPad * L[l].Kc * 2);
        Bp[l] = (float*)   alloc((size_t)L[l].CoutPad * 4);
    }
    _Float16* bufA = (_Float16*)alloc((size_t)M1 * 32 * 2);   //  64 MB
    _Float16* bufB = (_Float16*)alloc((size_t)M1 * 64 * 2);   // 128 MB

    for (int l = 0; l < 12; ++l) {
        int tot = L[l].CoutPad * L[l].Kc;
        prep_weight<<<gb(tot, 256), 256, 0, stream>>>(
            (const float*)d_in[L[l].widx], (const float*)d_in[L[l].widx + 1],
            Wh[l], Bp[l], L[l].Cout, L[l].Cin, L[l].Kc, tot);
    }

    auto run_gemm = [&](const _Float16* X, int l, _Float16* Yh, float* Yf,
                        int M, bool stats) {
        if (stats) {
            zero_f32<<<1, 256, 0, stream>>>(ssum, 256);
            zero_f32<<<1, 256, 0, stream>>>(ssq, 256);
        }
        int nt = L[l].CoutPad / 16, gy = 1;
        if (nt > 8) { gy = nt / 8; nt = 8; }
        dim3 grid((unsigned)(M / 16), (unsigned)gy);
        float* su = stats ? ssum : nullptr;
        float* sq = stats ? ssq : nullptr;
        const int Kc = L[l].Kc, Cp = L[l].CoutPad;
        switch (nt) {
        case 2: gemm_wmma_nt<2><<<grid, 32, 0, stream>>>(X, Wh[l], Bp[l], Yh, Yf, Kc, Cp, su, sq); break;
        case 3: gemm_wmma_nt<3><<<grid, 32, 0, stream>>>(X, Wh[l], Bp[l], Yh, Yf, Kc, Cp, su, sq); break;
        case 4: gemm_wmma_nt<4><<<grid, 32, 0, stream>>>(X, Wh[l], Bp[l], Yh, Yf, Kc, Cp, su, sq); break;
        default: gemm_wmma_nt<8><<<grid, 32, 0, stream>>>(X, Wh[l], Bp[l], Yh, Yf, Kc, Cp, su, sq); break;
        }
    };
    auto run_bn = [&](int l, float cnt) {
        bn_finalize<<<1, L[l].CoutPad, 0, stream>>>(
            ssum, ssq, (const float*)d_in[L[l].widx + 2],
            (const float*)d_in[L[l].widx + 3], scaleb, shiftb,
            L[l].Cout, 1.0f / cnt);
    };
    auto run_bnrelu = [&](_Float16* Y, int l, int M) {
        size_t tot = (size_t)M * L[l].CoutPad;
        bnrelu_kernel<<<gb(tot, 256), 256, 0, stream>>>(Y, scaleb, shiftb,
                                                        L[l].CoutPad, tot);
    };

    // ---- Stage 1: 1024 pts -> 128 centers, K=32, MLP 3->32->32->64 ----
    fps_kernel<<<256, 256, 0, stream>>>(pts, 1024, 128, 1024, 1, 3 * 1024, fidx1);
    gather_centers<<<gb(256 * 128 * 3, 256), 256, 0, stream>>>(
        pts, fidx1, nx1, 128, 1024, 1, 3 * 1024, 256 * 128 * 3);
    ball_query_kernel<<<256, 128, 0, stream>>>(pts, nx1, 1024, 128, 0.01f,
                                               1024, 1, 3 * 1024, idx1);
    gather_sa1<<<gb((size_t)M1 * 32, 256), 256, 0, stream>>>(pts, nx1, idx1,
                                                             bufA, M1 * 32);
    run_gemm(bufA, 0, bufB, nullptr, M1, true); run_bn(0, (float)M1); run_bnrelu(bufB, 0, M1);
    run_gemm(bufB, 1, bufA, nullptr, M1, true); run_bn(1, (float)M1); run_bnrelu(bufA, 1, M1);
    run_gemm(bufA, 2, bufB, nullptr, M1, true); run_bn(2, (float)M1);
    bnrelu_maxk<<<gb(256 * 128 * 64, 256), 256, 0, stream>>>(
        bufB, scaleb, shiftb, pooled1, 32, 64, 64, 256 * 128 * 64);

    // ---- Stage 2: 128 -> 32 centers, K=32, MLP 67->64->64->128 ----
    fps_kernel<<<256, 256, 0, stream>>>(nx1, 128, 32, 1, 3, 128 * 3, fidx2);
    gather_centers<<<gb(256 * 32 * 3, 256), 256, 0, stream>>>(
        nx1, fidx2, nx2, 32, 1, 3, 128 * 3, 256 * 32 * 3);
    ball_query_kernel<<<256, 128, 0, stream>>>(nx1, nx2, 128, 32, 0.04f,
                                               1, 3, 128 * 3, idx2);
    gather_sa2<<<gb((size_t)M2 * 96, 256), 256, 0, stream>>>(
        nx1, nx2, pooled1, idx2, bufA, M2 * 96);
    run_gemm(bufA, 3, bufB, nullptr, M2, true); run_bn(3, (float)M2); run_bnrelu(bufB, 3, M2);
    run_gemm(bufB, 4, bufA, nullptr, M2, true); run_bn(4, (float)M2); run_bnrelu(bufA, 4, M2);
    run_gemm(bufA, 5, bufB, nullptr, M2, true); run_bn(5, (float)M2);
    bnrelu_maxk<<<gb(256 * 32 * 128, 256), 256, 0, stream>>>(
        bufB, scaleb, shiftb, pooled2, 32, 128, 128, 256 * 32 * 128);

    // ---- Stage 3 (group_all): K=32, MLP 131->128->128->256 ----
    gather_sa3<<<gb((size_t)M3 * 160, 256), 256, 0, stream>>>(
        nx2, pooled2, bufA, M3 * 160);
    run_gemm(bufA, 6, bufB, nullptr, M3, true); run_bn(6, (float)M3); run_bnrelu(bufB, 6, M3);
    run_gemm(bufB, 7, bufA, nullptr, M3, true); run_bn(7, (float)M3); run_bnrelu(bufA, 7, M3);
    run_gemm(bufA, 8, bufB, nullptr, M3, true); run_bn(8, (float)M3);
    bnrelu_maxk<<<gb(256 * 256, 256), 256, 0, stream>>>(
        bufB, scaleb, shiftb, pooled3, 32, 256, 256, 256 * 256);

    // ---- FC head: 258->256->128->39 (BN over batch axis, cnt=256) ----
    build_fc_input<<<gb(256 * 288, 256), 256, 0, stream>>>(pooled3, onehot, fcX,
                                                           256 * 288);
    run_gemm(fcX, 9, bufA, nullptr, 256, true);  run_bn(9, 256.0f);  run_bnrelu(bufA, 9, 256);
    run_gemm(bufA, 10, bufB, nullptr, 256, true); run_bn(10, 256.0f); run_bnrelu(bufB, 10, 256);
    run_gemm(bufB, 11, nullptr, fcOut, 256, false);
    final_copy<<<gb(256 * 39, 256), 256, 0, stream>>>(fcOut, (float*)d_out, 256 * 39);
}
